// AtomAttentionBlock_44487271252534
// MI455X (gfx1250) — matrix operand order repaired and checked by reading, not compiled, not run on hardware
//
#include <hip/hip_runtime.h>
#include <hip/hip_bf16.h>

typedef __attribute__((ext_vector_type(8)))  _Float16 h8;
typedef __attribute__((ext_vector_type(16))) _Float16 v16h;
typedef __attribute__((ext_vector_type(8)))  float    v8f;

#define DEV __device__ __forceinline__

#if defined(__gfx1250__) && __has_builtin(__builtin_amdgcn_global_load_async_to_lds_b128)
#define USE_ASYNC 1
#else
#define USE_ASYNC 0
#endif

// Pointee/address-space types matching the builtin's signature:
// param0: int __attribute__((vector_size(16))) __device__(AS1) *
// param1: same vector type in LDS (AS3)
typedef int vi4 __attribute__((vector_size(16)));
typedef __attribute__((address_space(1))) vi4 gas_vi4;
typedef __attribute__((address_space(3))) vi4 las_vi4;

// Copy 16 bytes global -> LDS. Async path uses the CDNA5 ASYNCcnt-tracked
// direct-to-LDS instruction; fallback bounces through a VGPR (ds_store_b128).
DEV void cp_tile_16B(const _Float16* g, _Float16* l) {
#if USE_ASYNC
    __builtin_amdgcn_global_load_async_to_lds_b128((gas_vi4*)g, (las_vi4*)l, 0, 0);
#else
    *(h8*)l = *(const h8*)g;
#endif
}

DEV void async_fence() {
#if USE_ASYNC
#if __has_builtin(__builtin_amdgcn_s_wait_asynccnt)
    __builtin_amdgcn_s_wait_asynccnt(0);
#else
    asm volatile("s_wait_asynccnt 0" ::: "memory");
#endif
#endif
}

// Build a 16-half fragment from two contiguous 16-byte chunks (global or LDS).
DEV v16h frag16(const _Float16* p0, const _Float16* p1) {
    h8 a = *(const h8*)p0;
    h8 b = *(const h8*)p1;
    return __builtin_shufflevector(a, b, 0,1,2,3,4,5,6,7,8,9,10,11,12,13,14,15);
}

DEV v8f wmma_f16(v16h a, v16h b, v8f c) {
    // D = A(16x32 f16) * B(32x16 f16) + C(16x16 f32)
    return __builtin_amdgcn_wmma_f32_16x16x32_f16(false, a, false, b, (short)0, c,
                                                  false, false);
}

// ---------------------------------------------------------------------------
// f32 -> f16 conversion
// ---------------------------------------------------------------------------
__global__ void cvt_f16(const float* __restrict__ in, _Float16* __restrict__ out, int n) {
    int i = blockIdx.x * 256 + threadIdx.x;
    if (i < n) out[i] = (_Float16)in[i];
}

// ---------------------------------------------------------------------------
// LayerNorm over last dim C, one block (256 thr = 8 waves) per row; f16 out
// ---------------------------------------------------------------------------
__global__ void __launch_bounds__(256)
layernorm_k(const float* __restrict__ x, const float* __restrict__ g,
            const float* __restrict__ bta, _Float16* __restrict__ out, int C) {
    const int row = blockIdx.x;
    const int tid = threadIdx.x;
    const float* xr = x + (size_t)row * C;
    float s = 0.f, s2 = 0.f;
    for (int i = tid; i < C; i += 256) { float v = xr[i]; s += v; s2 += v * v; }
    for (int o = 16; o > 0; o >>= 1) { s += __shfl_down(s, o); s2 += __shfl_down(s2, o); }
    __shared__ float red[16];
    __shared__ float stat[2];
    if ((tid & 31) == 0) { red[tid >> 5] = s; red[8 + (tid >> 5)] = s2; }
    __syncthreads();
    if (tid == 0) {
        float S = 0.f, S2 = 0.f;
        for (int w2 = 0; w2 < 8; ++w2) { S += red[w2]; S2 += red[8 + w2]; }
        float mu = S / C;
        float var = S2 / C - mu * mu;
        stat[0] = mu;
        stat[1] = rsqrtf(var + 1e-5f);
    }
    __syncthreads();
    float mu = stat[0], rs = stat[1];
    for (int i = tid; i < C; i += 256)
        out[(size_t)row * C + i] = (_Float16)((xr[i] - mu) * rs * g[i] + bta[i]);
}

// ---------------------------------------------------------------------------
// WMMA GEMM: out = epilogue(A[M,K] @ Bw[N,K]^T).  Block = 8 waves, tile 128x64.
// B tile (64 rows x 32 k-halves = 4KB) is async-staged into LDS (shared by all
// 8 waves), double buffered; last K-step peeled so the steady-state loop has
// unconditional staging and batched ds_load fragments ahead of the 4 WMMAs.
// ---------------------------------------------------------------------------
__global__ void __launch_bounds__(256)
gemm_wmma(const _Float16* __restrict__ A, const _Float16* __restrict__ Bw,
          const float* __restrict__ bias, const float* __restrict__ res,
          _Float16* __restrict__ out16, float* __restrict__ out32,
          int M, int N, int K, float scale, int act) {
    __shared__ __align__(32) _Float16 bsm[2][64 * 32];     // 2 x 4KB
    const int tid  = threadIdx.x;
    const int lane = tid & 31;
    const int wv   = tid >> 5;
    const int lh   = lane & 15;
    const int sel  = lane >> 4;
    const int m0   = blockIdx.x * 128 + wv * 16;
    const int n0   = blockIdx.y * 64;

    // cooperative staging slot: row sj (0..63), 16B chunk sc
    const int sj = tid >> 2;
    const int sc = (tid & 3) * 8;                          // in halves
    const int sslot = sj * 32 + sc;
    const _Float16* gB = Bw + (size_t)(n0 + sj) * K + sc;

    cp_tile_16B(gB, &bsm[0][sslot]);                       // stage kb = 0

    v8f acc0 = {}, acc1 = {}, acc2 = {}, acc3 = {};
    const _Float16* arow = A + (size_t)(m0 + lh) * K;

    auto compute = [&](int kb, int bufi) {
        v16h a = frag16(arow + kb + 16 * sel, arow + kb + 16 * sel + 8);
        const _Float16* base = &bsm[bufi][lh * 32 + 16 * sel];
        // batch all 4 B-fragment ds_loads before issuing the WMMAs
        v16h bf0 = frag16(base,           base + 8);
        v16h bf1 = frag16(base + 16 * 32, base + 16 * 32 + 8);
        v16h bf2 = frag16(base + 32 * 32, base + 32 * 32 + 8);
        v16h bf3 = frag16(base + 48 * 32, base + 48 * 32 + 8);
        acc0 = wmma_f16(a, bf0, acc0);
        acc1 = wmma_f16(a, bf1, acc1);
        acc2 = wmma_f16(a, bf2, acc2);
        acc3 = wmma_f16(a, bf3, acc3);
    };

    int buf = 0;
    int kb  = 0;
    for (; kb < K - 32; kb += 32, buf ^= 1) {              // steady state
        async_fence();
        __syncthreads();                                   // staged data visible; WAR for restage
        cp_tile_16B(gB + kb + 32, &bsm[buf ^ 1][sslot]);   // unconditional prefetch
        compute(kb, buf);
    }
    async_fence();                                         // peeled final step
    __syncthreads();
    compute(kb, buf);

    v8f accs[4] = {acc0, acc1, acc2, acc3};
    for (int t = 0; t < 4; ++t) {
        int n = n0 + t * 16 + lh;
        float bv = bias ? bias[n] : 0.0f;
        for (int r = 0; r < 8; ++r) {
            int m = m0 + r + 8 * sel;
            float v = (accs[t][r] + bv) * scale;
            if (act) v = 0.5f * v * (1.0f + erff(v * 0.70710678118f));  // exact GELU
            size_t idx = (size_t)m * N + n;
            if (res)   v += res[idx];
            if (out16) out16[idx] = (_Float16)v;
            if (out32) out32[idx] = v;
        }
    }
}

// ---------------------------------------------------------------------------
// V transpose: v[b,s,h*64+d] -> vT[b,h,d,s]  (so PV B-fragments are contiguous)
// ---------------------------------------------------------------------------
__global__ void transpose_v(const _Float16* __restrict__ v, _Float16* __restrict__ vT) {
    constexpr int S = 2048, C = 512, H = 8, D = 64;
    int i = blockIdx.x * 256 + threadIdx.x;
    if (i >= 2 * S * C) return;
    int c = i & (C - 1);
    int s = (i / C) & (S - 1);
    int b = i / (C * S);
    int h = c >> 6, d = c & 63;
    vT[(((size_t)(b * H + h)) * D + d) * S + s] = v[i];
}

// ---------------------------------------------------------------------------
// Flash attention: block = 8 consecutive 16-query tiles of one (b, h); the
// 32-key K-chunk (32x64) and V-chunk (64x32, from pre-transposed vT) are
// async-staged into LDS once per block per step (8x VMEM reduction), double
// buffered, last chunk peeled. Online softmax in registers; probability tile
// bounced through per-wave LDS to form the PV A-fragment.
// ---------------------------------------------------------------------------
__global__ void __launch_bounds__(256)
attention_k(const _Float16* __restrict__ q, const _Float16* __restrict__ k,
            const _Float16* __restrict__ vT, const float* __restrict__ pb,
            const unsigned char* __restrict__ mask, _Float16* __restrict__ out) {
    constexpr int S = 2048, C = 512, H = 8, D = 64, L = 512;
    __shared__ __align__(32) _Float16 ksm[2][32 * 64];     // [key][d]  2 x 4KB
    __shared__ __align__(32) _Float16 vsm[2][64 * 32];     // [d][key]  2 x 4KB
    __shared__ __align__(32) _Float16 pbuf[8][16 * 32];    // per-wave P tile, 8KB

    const int tid  = threadIdx.x;
    const int lane = tid & 31;
    const int wv   = tid >> 5;
    const int lh   = lane & 15;
    const int sel  = lane >> 4;

    const int qg = blockIdx.x & 15;            // 16 query groups per (b,h)
    const int h  = (blockIdx.x >> 4) & 7;
    const int b  = blockIdx.x >> 7;
    const int q0 = (qg * 8 + wv) * 16;

    // staging slots
    const int kj = tid >> 3;                   // K-tile row 0..31 (key)
    const int kc = (tid & 7) * 8;              // halves within 64-wide row
    const int vj = tid >> 2;                   // V-tile row 0..63 (d)
    const int vc = (tid & 3) * 8;              // halves within 32-wide row
    const int kslot = kj * 64 + kc;
    const int vslot = vj * 32 + vc;
    const _Float16* gK = k + (size_t)(b * S + kj) * C + h * D + kc;
    const _Float16* gV = vT + ((size_t)(b * H + h) * D + vj) * S + vc;

    // Preload Q fragments (16 rows x 64 d, two K=32 chunks)
    const _Float16* qrow = q + (size_t)(b * S + q0 + lh) * C + h * D;
    v16h aq0 = frag16(qrow + 16 * sel,      qrow + 16 * sel + 8);
    v16h aq1 = frag16(qrow + 32 + 16 * sel, qrow + 40 + 16 * sel);

    // Periodic pair bias is invariant over the key loop for this lane/row.
    float brow[8];
    for (int r = 0; r < 8; ++r) brow[r] = pb[(h * 4 + (r & 3)) * 4 + (lh & 3)];

    float mrun[8], lrun[8];
    for (int r = 0; r < 8; ++r) { mrun[r] = -1e30f; lrun[r] = 0.f; }
    v8f o0 = {}, o1 = {}, o2 = {}, o3 = {};

    _Float16* pl = &pbuf[wv][0];

    auto process = [&](int kt, int bufi) {
        const unsigned char msk0 = mask[b * L + ((kt + lh) >> 2)];
        const unsigned char msk1 = mask[b * L + ((kt + 16 + lh) >> 2)];

        // scores: Q[16x64] @ K^T -> two 16x16 tiles (K frags from LDS, batched)
        const _Float16* kr0 = &ksm[bufi][(lh)      * 64 + 16 * sel];
        const _Float16* kr1 = &ksm[bufi][(16 + lh) * 64 + 16 * sel];
        v16h bk00 = frag16(kr0,      kr0 + 8);
        v16h bk01 = frag16(kr0 + 32, kr0 + 40);
        v16h bk10 = frag16(kr1,      kr1 + 8);
        v16h bk11 = frag16(kr1 + 32, kr1 + 40);
        v8f s0 = {}, s1 = {};
        s0 = wmma_f16(aq0, bk00, s0);
        s0 = wmma_f16(aq1, bk01, s0);
        s1 = wmma_f16(aq0, bk10, s1);
        s1 = wmma_f16(aq1, bk11, s1);

        // online softmax over this 32-key chunk (row reductions within 16-lane half)
        for (int r = 0; r < 8; ++r) {
            float v0 = msk0 ? (s0[r] + brow[r]) : -1e30f;
            float v1 = msk1 ? (s1[r] + brow[r]) : -1e30f;
            float rm = fmaxf(v0, v1);
            rm = fmaxf(rm, __shfl_xor(rm, 1));
            rm = fmaxf(rm, __shfl_xor(rm, 2));
            rm = fmaxf(rm, __shfl_xor(rm, 4));
            rm = fmaxf(rm, __shfl_xor(rm, 8));
            float nm   = fmaxf(mrun[r], rm);
            float corr = __expf(mrun[r] - nm);
            float p0 = __expf(v0 - nm);
            float p1 = __expf(v1 - nm);
            float rs = p0 + p1;
            rs += __shfl_xor(rs, 1);
            rs += __shfl_xor(rs, 2);
            rs += __shfl_xor(rs, 4);
            rs += __shfl_xor(rs, 8);
            lrun[r] = lrun[r] * corr + rs;
            mrun[r] = nm;
            o0[r] *= corr; o1[r] *= corr; o2[r] *= corr; o3[r] *= corr;
            // C/D layout -> per-wave LDS (row-major 16x32 f16 tile)
            pl[(r + 8 * sel) * 32 + lh]      = (_Float16)p0;
            pl[(r + 8 * sel) * 32 + 16 + lh] = (_Float16)p1;
        }

        // Re-read P in A-fragment layout (same-wave LDS RAW is in-order)
        const _Float16* prow = pl + lh * 32;
        v16h ap = frag16(prow + 16 * sel, prow + 16 * sel + 8);

        // O += P[16x32] @ V[32x64] : 4 N-tiles of 16 along d (V frags, batched)
        const _Float16* vbase = &vsm[bufi][lh * 32 + 16 * sel];
        v16h bv0 = frag16(vbase,           vbase + 8);
        v16h bv1 = frag16(vbase + 16 * 32, vbase + 16 * 32 + 8);
        v16h bv2 = frag16(vbase + 32 * 32, vbase + 32 * 32 + 8);
        v16h bv3 = frag16(vbase + 48 * 32, vbase + 48 * 32 + 8);
        o0 = wmma_f16(ap, bv0, o0);
        o1 = wmma_f16(ap, bv1, o1);
        o2 = wmma_f16(ap, bv2, o2);
        o3 = wmma_f16(ap, bv3, o3);
    };

    // stage kt = 0 into buffer 0
    cp_tile_16B(gK, &ksm[0][kslot]);
    cp_tile_16B(gV, &vsm[0][vslot]);

    int buf = 0;
    int kt  = 0;
    for (; kt < S - 32; kt += 32, buf ^= 1) {              // steady state
        async_fence();
        __syncthreads();
        cp_tile_16B(gK + (size_t)(kt + 32) * C, &ksm[buf ^ 1][kslot]);
        cp_tile_16B(gV + (kt + 32),             &vsm[buf ^ 1][vslot]);
        process(kt, buf);
    }
    async_fence();                                         // peeled final chunk
    __syncthreads();
    process(kt, buf);

    v8f oacc[4] = {o0, o1, o2, o3};
    for (int t = 0; t < 4; ++t)
        for (int r = 0; r < 8; ++r) {
            int m = q0 + r + 8 * sel;
            float val = oacc[t][r] / (lrun[r] + 1e-20f);
            out[(size_t)(b * S + m) * C + h * D + t * 16 + lh] = (_Float16)val;
        }
}

// ---------------------------------------------------------------------------
extern "C" void kernel_launch(void* const* d_in, const int* in_sizes, int n_in,
                              void* d_out, int out_size, void* d_ws, size_t ws_size,
                              hipStream_t stream) {
    (void)in_sizes; (void)n_in; (void)out_size; (void)ws_size;

    const float* x     = (const float*)d_in[0];
    const unsigned char* mask = (const unsigned char*)d_in[1];
    const float* ln1_g = (const float*)d_in[2];
    const float* ln1_b = (const float*)d_in[3];
    const float* Wq = (const float*)d_in[4];
    const float* bq = (const float*)d_in[5];
    const float* Wk = (const float*)d_in[6];
    const float* bk = (const float*)d_in[7];
    const float* Wv = (const float*)d_in[8];
    const float* bv = (const float*)d_in[9];
    const float* Wo = (const float*)d_in[10];
    const float* bo = (const float*)d_in[11];
    const float* pb = (const float*)d_in[12];
    const float* ln2_g = (const float*)d_in[13];
    const float* ln2_b = (const float*)d_in[14];
    const float* W1 = (const float*)d_in[15];
    const float* b1 = (const float*)d_in[16];
    const float* W2 = (const float*)d_in[17];
    const float* b2 = (const float*)d_in[18];
    float* out = (float*)d_out;

    constexpr int Bn = 2, S = 2048, C = 512;
    constexpr int M = Bn * S;                     // 4096 rows

    char* w = (char*)d_ws;
    auto MB = [](size_t v) { return v << 20; };
    _Float16* h16  = (_Float16*)(w);              // LN1 output, f16      4 MB
    _Float16* q16  = (_Float16*)(w + MB(4));
    _Float16* k16  = (_Float16*)(w + MB(8));
    _Float16* v16  = (_Float16*)(w + MB(12));
    _Float16* vT16 = (_Float16*)(w + MB(16));
    _Float16* ao16 = (_Float16*)(w + MB(20));     // attention output
    float*    x1   = (float*)   (w + MB(24));     // residual 1, f32      8 MB
    _Float16* h216 = (_Float16*)(w + MB(32));     // LN2 output
    _Float16* ff16 = (_Float16*)(w + MB(36));     // FFN hidden          16 MB
    _Float16* Wq16 = (_Float16*)(w + MB(52));
    _Float16* Wk16 = (_Float16*)(w + MB(52) + (512 << 10));
    _Float16* Wv16 = (_Float16*)(w + MB(53));
    _Float16* Wo16 = (_Float16*)(w + MB(53) + (512 << 10));
    _Float16* W116 = (_Float16*)(w + MB(54));
    _Float16* W216 = (_Float16*)(w + MB(56));

    // weight f32 -> f16
    const int nw = C * C;
    cvt_f16<<<(nw + 255) / 256, 256, 0, stream>>>(Wq, Wq16, nw);
    cvt_f16<<<(nw + 255) / 256, 256, 0, stream>>>(Wk, Wk16, nw);
    cvt_f16<<<(nw + 255) / 256, 256, 0, stream>>>(Wv, Wv16, nw);
    cvt_f16<<<(nw + 255) / 256, 256, 0, stream>>>(Wo, Wo16, nw);
    const int nw1 = 4 * C * C;
    cvt_f16<<<(nw1 + 255) / 256, 256, 0, stream>>>(W1, W116, nw1);
    cvt_f16<<<(nw1 + 255) / 256, 256, 0, stream>>>(W2, W216, nw1);

    // LN1
    layernorm_k<<<M, 256, 0, stream>>>(x, ln1_g, ln1_b, h16, C);

    // QKV projections (q pre-scaled by 1/sqrt(64) = 0.125)
    dim3 g1(M / 128, C / 64);
    gemm_wmma<<<g1, 256, 0, stream>>>(h16, Wq16, bq, nullptr, q16, nullptr, M, C, C, 0.125f, 0);
    gemm_wmma<<<g1, 256, 0, stream>>>(h16, Wk16, bk, nullptr, k16, nullptr, M, C, C, 1.0f, 0);
    gemm_wmma<<<g1, 256, 0, stream>>>(h16, Wv16, bv, nullptr, v16, nullptr, M, C, C, 1.0f, 0);

    // V -> [b,h,d,s]
    const int nt = M * C;
    transpose_v<<<(nt + 255) / 256, 256, 0, stream>>>(v16, vT16);

    // flash attention: 256 blocks x 8 waves; block = 8 query tiles of one (b,h)
    attention_k<<<256, 256, 0, stream>>>(q16, k16, vT16, pb, mask, ao16);

    // O projection + residual -> x1 (f32)
    gemm_wmma<<<g1, 256, 0, stream>>>(ao16, Wo16, bo, x, nullptr, x1, M, C, C, 1.0f, 0);

    // LN2
    layernorm_k<<<M, 256, 0, stream>>>(x1, ln2_g, ln2_b, h216, C);

    // FFN1 + exact GELU
    dim3 g2(M / 128, (4 * C) / 64);
    gemm_wmma<<<g2, 256, 0, stream>>>(h216, W116, b1, nullptr, ff16, nullptr, M, 4 * C, C, 1.0f, 1);

    // FFN2 + residual -> out (f32)
    dim3 g3(M / 128, C / 64);
    gemm_wmma<<<g3, 256, 0, stream>>>(ff16, W216, b2, x1, nullptr, out, M, C, 4 * C, 1.0f, 0);
}
